// MatchTensorScore_24575802867777
// MI455X (gfx1250) — compile-verified, matrix operand order checked by reading, not run on hardware
//
#include <hip/hip_runtime.h>
#include <stdint.h>

// ---------------------------------------------------------------------------
// Problem constants
// ---------------------------------------------------------------------------
#define E_DIM   1024
#define R_DIM   32
#define B_DIM   4096
#define F_CHUNK 64                    // f values per block  -> grid.y = 16

// bf16 / f32 vector types for WMMA
typedef __bf16 v16bf __attribute__((ext_vector_type(16)));
typedef float  v8f   __attribute__((ext_vector_type(8)));

#define WMMA_BF16(a, b, c) \
  __builtin_amdgcn_wmma_f32_16x16x32_bf16(false, (a), false, (b), (short)0, (c), false, false)

// ---------------------------------------------------------------------------
// Workspace layout (bytes)
// ---------------------------------------------------------------------------
#define OFF_MP   ((size_t)0)
#define OFF_X1P  (OFF_MP  + (size_t)33554432 * 2)
#define OFF_X2P  (OFF_X1P + (size_t)4194304 * 2)
#define OFF_VP   (OFF_X2P + (size_t)4194304 * 2)
#define OFF_P1   (OFF_VP  + (size_t)65536   * 2)

// ---------------------------------------------------------------------------
// Helpers
// ---------------------------------------------------------------------------
__device__ __forceinline__ unsigned short f32_to_bf16(float f) {
  unsigned int u = __float_as_uint(f);
  u += 0x7FFFu + ((u >> 16) & 1u);          // round-to-nearest-even
  return (unsigned short)(u >> 16);
}

// K index within a 32-wide K-block -> (lane-half, element) per CDNA5 16-bit
// fragment layout (ISA 7.12.2): half0 lanes hold K=0..7,16..23; half1 the rest.
__device__ __forceinline__ void kmap(int k, int& half, int& eidx) {
  if (k < 16) { half = k >> 3;        eidx = k & 7; }
  else        { half = (k - 16) >> 3; eidx = 8 + ((k - 16) & 7); }
}

// ---------------------------------------------------------------------------
// Pack M (E x E*R f32) -> bf16 B-matrix fragments, frag id = ntile*32+kblock,
// 512 bf16 per frag, lane-major (16 per lane).  N = f*32 + r.
// ---------------------------------------------------------------------------
__global__ void pack_M_kernel(const float* __restrict__ M,
                              unsigned short* __restrict__ Mp) {
  int idx = blockIdx.x * 256 + threadIdx.x;          // 0 .. 33554431
  int e = idx >> 15;
  int n = idx & 32767;
  unsigned short v = f32_to_bf16(M[idx]);
  int t  = n >> 4, nl = n & 15;
  int kb = e >> 5, k  = e & 31;
  int half, eidx; kmap(k, half, eidx);
  size_t dst = ((size_t)(t * 32 + kb) << 9) + (size_t)((half * 16 + nl) << 4) + eidx;
  Mp[dst] = v;
}

__global__ void pack_X_kernel(const float* __restrict__ X,
                              unsigned short* __restrict__ Xp) {
  int idx = blockIdx.x * 256 + threadIdx.x;          // 0 .. 4194303
  int b = idx >> 10;
  int e = idx & 1023;
  unsigned short v = f32_to_bf16(X[idx]);
  int rg = b >> 4, m = b & 15;
  int kb = e >> 5, k = e & 31;
  int half, eidx; kmap(k, half, eidx);
  size_t dst = ((size_t)(rg * 32 + kb) << 9) + (size_t)((half * 16 + m) << 4) + eidx;
  Xp[dst] = v;
}

__global__ void pack_V_kernel(const float* __restrict__ V,
                              unsigned short* __restrict__ Vp) {
  int idx = blockIdx.x * 256 + threadIdx.x;          // 0 .. 65535
  int r = idx >> 11;
  int c = idx & 2047;
  int p = c >> 10;
  int e = c & 1023;
  unsigned short v = f32_to_bf16(V[idx]);
  int nt = r >> 4, nl = r & 15;
  int kb = e >> 5, k = e & 31;
  int half, eidx; kmap(k, half, eidx);
  size_t dst = ((size_t)((p * 2 + nt) * 32 + kb) << 9) +
               (size_t)((half * 16 + nl) << 4) + eidx;
  Vp[dst] = v;
}

__global__ void zero_p1_kernel(float* __restrict__ p1) {
  int idx = blockIdx.x * 256 + threadIdx.x;
  p1[idx] = 0.0f;
}

// ---------------------------------------------------------------------------
// Async staging: copy the 64KB fragment block for one f value
// (64 contiguous frags = Mp[f*65536B .. +64KB)) into LDS at ldsDst.
// 128 threads x 32 x b128 -> ASYNCcnt-tracked, no VGPR round trip.
// ---------------------------------------------------------------------------
__device__ __forceinline__ void stage_M_async(const unsigned short* __restrict__ Mp,
                                              int f, unsigned ldsDst, int tid) {
  const char* g0 = (const char*)Mp + ((size_t)f << 16) + ((size_t)tid << 4);
  unsigned l0 = ldsDst + ((unsigned)tid << 4);
#pragma unroll
  for (int i = 0; i < 32; ++i) {
    asm volatile("global_load_async_to_lds_b128 %0, %1, off"
                 :: "v"(l0 + ((unsigned)i << 11)),
                    "v"((unsigned long long)(g0 + ((size_t)i << 11)))
                 : "memory");
  }
}

// ---------------------------------------------------------------------------
// Main fused bilinear kernel.
//   grid = (B/128, 1024/F_CHUNK), block = 128 (4 waves).
//   Each wave: 32 batch rows (two 16-row tiles) x all 32 r (two N-tiles).
//   M fragments are async-staged into LDS once per block (4x L2 cut),
//   double-buffered across f.  A fragments load direct (row-private).
// ---------------------------------------------------------------------------
__global__ __launch_bounds__(128, 1)
void bilinear_wmma_kernel(const unsigned short* __restrict__ Mp,
                          const unsigned short* __restrict__ X1p,
                          const unsigned short* __restrict__ X2p,
                          const unsigned short* __restrict__ Vp,
                          const float* __restrict__ x2,
                          float* __restrict__ p1) {
  __shared__ unsigned short Mstage[2][32768];        // 2 x 64KB double buffer

  const int tid      = threadIdx.x;
  const int lane     = tid & 31;
  const int wave     = tid >> 5;
  const int rowbase  = blockIdx.x * 128 + wave * 32;
  const int rg0      = rowbase >> 4;
  const int laneElem = lane << 4;                    // 16 bf16 per lane per frag
  const int f0       = blockIdx.y * F_CHUNK;
  const int half8    = (lane >> 4) << 3;             // C-layout row offset
  const int nl       = lane & 15;                    // C-layout column

  const unsigned ldsRoot = (unsigned)(uintptr_t)&Mstage[0][0];

  v8f acc00 = {}, acc01 = {}, acc10 = {}, acc11 = {};

  const unsigned short* ab = X1p + (((size_t)rg0 * 32) << 9);

  // prologue: stage f0 into buffer 0
  stage_M_async(Mp, f0, ldsRoot, tid);

  for (int f = f0; f < f0 + F_CHUNK; ++f) {
    const int cur = (f - f0) & 1;
    // stage next f into the other buffer, then wait for the current one
    if (f + 1 < f0 + F_CHUNK) {
      stage_M_async(Mp, f + 1, ldsRoot + (unsigned)((1 - cur) << 16), tid);
      asm volatile("s_wait_asynccnt 0x20" ::: "memory");   // current batch done
    } else {
      asm volatile("s_wait_asynccnt 0x0" ::: "memory");
    }
    __syncthreads();                                       // publish staged data

    const v16bf* mb = (const v16bf*)&Mstage[cur][0];       // 32B units
    v8f g00 = {}, g01 = {}, g10 = {}, g11 = {};
#pragma unroll 4
    for (int kb = 0; kb < 32; ++kb) {
      v16bf a0 = *(const v16bf*)(ab + ((size_t)kb << 9)        + laneElem);
      v16bf a1 = *(const v16bf*)(ab + ((size_t)(32 + kb) << 9) + laneElem);
      v16bf b0 = mb[(kb << 5) + lane];                     // nt = 0
      v16bf b1 = mb[((32 + kb) << 5) + lane];              // nt = 1
      g00 = WMMA_BF16(a0, b0, g00);
      g01 = WMMA_BF16(a0, b1, g01);
      g10 = WMMA_BF16(a1, b0, g10);
      g11 = WMMA_BF16(a1, b1, g11);
    }
    __syncthreads();                                       // reads done before overwrite

    // part1: scale G rows by x2[row, f]
    const float* x2r = x2 + (size_t)(rowbase + half8) * E_DIM + f;
#pragma unroll
    for (int j = 0; j < 8; ++j) {
      float s0 = x2r[(size_t)j * E_DIM];
      float s1 = x2r[(size_t)(j + 16) * E_DIM];
      acc00[j] += s0 * g00[j];
      acc01[j] += s0 * g01[j];
      acc10[j] += s1 * g10[j];
      acc11[j] += s1 * g11[j];
    }
  }

  if (blockIdx.y == 0) {
    // part2 = x1 @ V[:, :E].T + x2 @ V[:, E:].T  folded straight into acc
#pragma unroll 1
    for (int pass = 0; pass < 2; ++pass) {
      const unsigned short* ap = (pass ? X2p : X1p) + (((size_t)rg0 * 32) << 9);
      const unsigned short* vb = Vp + (((size_t)(pass * 2) * 32) << 9);
#pragma unroll 4
      for (int kb = 0; kb < 32; ++kb) {
        v16bf a0 = *(const v16bf*)(ap + ((size_t)kb << 9)        + laneElem);
        v16bf a1 = *(const v16bf*)(ap + ((size_t)(32 + kb) << 9) + laneElem);
        v16bf b0 = *(const v16bf*)(vb + ((size_t)kb << 9)        + laneElem);
        v16bf b1 = *(const v16bf*)(vb + ((size_t)(32 + kb) << 9) + laneElem);
        acc00 = WMMA_BF16(a0, b0, acc00);
        acc01 = WMMA_BF16(a0, b1, acc01);
        acc10 = WMMA_BF16(a1, b0, acc10);
        acc11 = WMMA_BF16(a1, b1, acc11);
      }
    }
  }

  // scatter partials: C layout -> row = base + half8 + j, col = nl (+16)
#pragma unroll
  for (int j = 0; j < 8; ++j) {
    int r0 = rowbase + half8 + j;
    int r1 = r0 + 16;
    atomicAdd(&p1[r0 * R_DIM + nl],      acc00[j]);
    atomicAdd(&p1[r0 * R_DIM + 16 + nl], acc01[j]);
    atomicAdd(&p1[r1 * R_DIM + nl],      acc10[j]);
    atomicAdd(&p1[r1 * R_DIM + 16 + nl], acc11[j]);
  }
}

// ---------------------------------------------------------------------------
// Finalize: out[b] = sum_r u[r] * tanh(p1[b,r] + bias[r])
// ---------------------------------------------------------------------------
__global__ void finalize_kernel(const float* __restrict__ p1,
                                const float* __restrict__ bias,
                                const float* __restrict__ u,
                                float* __restrict__ out) {
  int lane = threadIdx.x & 31;
  int wave = threadIdx.x >> 5;
  int b = blockIdx.x * 8 + wave;
  float v = tanhf(p1[b * R_DIM + lane] + bias[lane]) * u[lane];
#pragma unroll
  for (int off = 16; off > 0; off >>= 1)
    v += __shfl_xor(v, off, 32);
  if (lane == 0) out[b] = v;
}

// ---------------------------------------------------------------------------
// Launch
// ---------------------------------------------------------------------------
extern "C" void kernel_launch(void* const* d_in, const int* in_sizes, int n_in,
                              void* d_out, int out_size, void* d_ws, size_t ws_size,
                              hipStream_t stream) {
  const float* x1 = (const float*)d_in[0];   // [4096,1024]
  const float* x2 = (const float*)d_in[1];   // [4096,1024]
  const float* M  = (const float*)d_in[2];   // [1024,1024,32]
  const float* V  = (const float*)d_in[3];   // [32,2048]
  const float* bb = (const float*)d_in[4];   // [1,32]
  const float* u  = (const float*)d_in[5];   // [1,32]
  float* out = (float*)d_out;                // [4096,1]

  char* ws = (char*)d_ws;
  unsigned short* Mp  = (unsigned short*)(ws + OFF_MP);
  unsigned short* X1p = (unsigned short*)(ws + OFF_X1P);
  unsigned short* X2p = (unsigned short*)(ws + OFF_X2P);
  unsigned short* Vp  = (unsigned short*)(ws + OFF_VP);
  float*          p1  = (float*)(ws + OFF_P1);

  // 1) convert + fragment-swizzle operands to bf16
  pack_M_kernel<<<33554432 / 256, 256, 0, stream>>>(M, Mp);
  pack_X_kernel<<<4194304 / 256, 256, 0, stream>>>(x1, X1p);
  pack_X_kernel<<<4194304 / 256, 256, 0, stream>>>(x2, X2p);
  pack_V_kernel<<<65536 / 256, 256, 0, stream>>>(V, Vp);

  // 2) zero the accumulator (atomics below; must re-zero every launch)
  zero_p1_kernel<<<(B_DIM * R_DIM) / 256, 256, 0, stream>>>(p1);

  // 3) fused bilinear + part2 GEMMs (WMMA bf16, async-LDS staged M)
  dim3 grid(B_DIM / 128, E_DIM / F_CHUNK);
  bilinear_wmma_kernel<<<grid, 128, 0, stream>>>(Mp, X1p, X2p, Vp, x2, p1);

  // 4) tanh + u-projection
  finalize_kernel<<<B_DIM / 8, 256, 0, stream>>>(p1, bb, u, out);
}
